// Model_73770358276764
// MI455X (gfx1250) — matrix operand order repaired and checked by reading, not compiled, not run on hardware
//
#include <hip/hip_runtime.h>
#include <math.h>

#define ENT_N 14541
#define REL_N 237
#define DIM   50

typedef __attribute__((ext_vector_type(16))) _Float16 v16h;
typedef __attribute__((ext_vector_type(8)))  float    v8f;

// ---- workspace layout ----
// ints: [0,256) counts | [256,512) offsets | [512,768) cursors | [768,768+n) perm
// then (128B-aligned): xp = f16 X matrix, n rows x 64 halves (128 B each),
// row i = f16(h-t) of the item at perm[i], K zero-padded 50..63.

__global__ void zero_k(int* ws) {
    int i = blockIdx.x * blockDim.x + threadIdx.x;
    if (i < 768) ws[i] = 0;
}

__global__ void hist_k(const int* __restrict__ r_typ, int* __restrict__ counts, int n) {
    int i = blockIdx.x * blockDim.x + threadIdx.x;
    if (i < n) atomicAdd(&counts[r_typ[i]], 1);
}

__global__ void scan_k(int* ws) {
    if (threadIdx.x == 0 && blockIdx.x == 0) {
        int acc = 0;
        for (int i = 0; i < REL_N; ++i) {
            ws[256 + i] = acc;   // offsets
            ws[512 + i] = acc;   // cursors
            acc += ws[i];
        }
    }
}

__global__ void scatter_k(const int* __restrict__ r_typ, int* __restrict__ cursors,
                          int* __restrict__ perm, int n) {
    int i = blockIdx.x * blockDim.x + threadIdx.x;
    if (i < n) {
        int pos = atomicAdd(&cursors[r_typ[i]], 1);
        perm[pos] = i;
    }
}

__device__ __forceinline__ int iclamp(int v, int hi) { return v > hi ? hi : v; }

// Coalesced staging pass: xp[pos][0..63] = f16(ent[h]-ent[t]) for perm[pos],
// zero-padded K=50..63. One thread per packed dword (2 halves).
__global__ void xprep_k(const float* __restrict__ ent, const int* __restrict__ h_ids,
                        const int* __restrict__ t_ids, const int* __restrict__ perm,
                        unsigned* __restrict__ xp32, int n) {
    int j = blockIdx.x * blockDim.x + threadIdx.x;   // dword index: n*32 total
    if (j >= n * 32) return;
    const int pos = j >> 5;
    const int k0  = (j & 31) * 2;                    // even, 0..62
    const int b   = perm[pos];
    const float* hrow = ent + h_ids[b] * DIM;
    const float* trow = ent + t_ids[b] * DIM;
    const int  k0c = iclamp(k0, DIM - 2);            // <=48, elems 48/49 legal
    const bool ok  = k0 < DIM;
    float2 xh = *(const float2*)(hrow + k0c);
    float2 xt = *(const float2*)(trow + k0c);
    union { _Float16 h[2]; unsigned u; } pk;
    pk.h[0] = (_Float16)(ok ? (xh.x - xt.x) : 0.0f);
    pk.h[1] = (_Float16)(ok ? (xh.y - xt.y) : 0.0f);
    xp32[j] = pk.u;
}

// Shared helper: preload A fragments (M_r zero-padded to 64x64 in f16) and the
// lane's 32 relation components (zeroed for rows >= 50 so the reduction is maskless).
// A k order used here (contiguous, B uses the same): lane row = l16 (+16m), VGPR v holds
// K = 2*v + 16*half (+1).
__device__ __forceinline__ void load_a_rr(const float* __restrict__ mat,
                                          const float* __restrict__ rel,
                                          int r, int half, int l16,
                                          v16h a[4][2], float rr[4][8]) {
    const float* Mb = mat + r * (DIM * DIM);
    #pragma unroll
    for (int m = 0; m < 4; ++m) {
        const int row  = 16 * m + l16;
        const int rowc = iclamp(row, DIM - 1);
        const bool rok = row < DIM;
        #pragma unroll
        for (int kc = 0; kc < 2; ++kc) {
            v16h frag;
            #pragma unroll
            for (int v = 0; v < 8; ++v) {
                const int k0  = 2 * v + 16 * half + 32 * kc;
                const int k0c = iclamp(k0, DIM - 2);
                const bool ok = rok && (k0 < DIM);
                float2 x = *(const float2*)(Mb + rowc * DIM + k0c);
                frag[2 * v]     = (_Float16)(ok ? x.x : 0.0f);
                frag[2 * v + 1] = (_Float16)(ok ? x.y : 0.0f);
            }
            a[m][kc] = frag;
        }
    }
    const float* rrow = rel + r * DIM;
    #pragma unroll
    for (int m = 0; m < 4; ++m)
        #pragma unroll
        for (int v = 0; v < 8; ++v) {
            const int row = 16 * m + v + 8 * half;
            const float rv = rrow[iclamp(row, DIM - 1)];
            rr[m][v] = (row < DIM) ? rv : 0.0f;      // masked ONCE, loop is maskless
        }
}

__device__ __forceinline__ void wmma_tail(const v16h a[4][2], const v16h bfrag[2],
                                          const float rr[4][8], int half,
                                          bool valid, int b, float* __restrict__ out) {
    v8f acc[4];
    #pragma unroll
    for (int m = 0; m < 4; ++m) {
        v8f z = {};
        v8f c0 = __builtin_amdgcn_wmma_f32_16x16x32_f16(
            false, a[m][0], false, bfrag[0], (short)0, z, false, false);
        acc[m] = __builtin_amdgcn_wmma_f32_16x16x32_f16(
            false, a[m][1], false, bfrag[1], (short)0, c0, false, false);
    }
    // maskless: A rows>=50 are zero => acc rows>=50 are zero; rr zeroed there too.
    float s = 0.f;
    #pragma unroll
    for (int m = 0; m < 4; ++m)
        #pragma unroll
        for (int v = 0; v < 8; ++v) {
            const float d = acc[m][v] + rr[m][v];
            s = fmaf(d, d, s);
        }
    s += __shfl_xor(s, 16, 32);
    if (half == 0 && valid) out[b] = sqrtf(s);
}

// Staged GEMM: B fragments are two contiguous 32-byte v16h loads from xp.
__global__ __launch_bounds__(32) void transr_wmma_staged_k(
    const float* __restrict__ rel, const float* __restrict__ mat,
    const int* __restrict__ perm, const int* __restrict__ counts,
    const int* __restrict__ offsets, const _Float16* __restrict__ xp,
    float* __restrict__ out)
{
    const int r     = blockIdx.x;
    const int half  = threadIdx.x >> 4;
    const int l16   = threadIdx.x & 15;
    const int n_r   = counts[r];
    const int start = offsets[r];
    const int ntile = (n_r + 15) >> 4;

    v16h a[4][2]; float rr[4][8];
    load_a_rr(mat, rel, r, half, l16, a, rr);

    for (int tile = blockIdx.y; tile < ntile; tile += gridDim.y) {
        const int  cidx  = tile * 16 + l16;
        const bool valid = cidx < n_r;
        const int  rowi  = start + (valid ? cidx : 0);
        const int  b     = perm[rowi];
        const _Float16* xr = xp + (size_t)rowi * 64;
        v16h bfrag[2];
        bfrag[0] = *(const v16h*)(xr + 16 * half);        // K 0..31 half
        bfrag[1] = *(const v16h*)(xr + 32 + 16 * half);   // K 32..63 half
        wmma_tail(a, bfrag, rr, half, valid, b, out);
    }
}

// Direct-gather fallback (used when ws is too small for xp). B tail needs no
// masking: A is zero for K>=50, and clamped loads are finite.
__global__ __launch_bounds__(32) void transr_wmma_direct_k(
    const float* __restrict__ ent, const float* __restrict__ rel,
    const float* __restrict__ mat, const int* __restrict__ h_ids,
    const int* __restrict__ t_ids, const int* __restrict__ perm,
    const int* __restrict__ counts, const int* __restrict__ offsets,
    float* __restrict__ out)
{
    const int r     = blockIdx.x;
    const int half  = threadIdx.x >> 4;
    const int l16   = threadIdx.x & 15;
    const int n_r   = counts[r];
    const int start = offsets[r];
    const int ntile = (n_r + 15) >> 4;

    v16h a[4][2]; float rr[4][8];
    load_a_rr(mat, rel, r, half, l16, a, rr);

    for (int tile = blockIdx.y; tile < ntile; tile += gridDim.y) {
        const int  cidx  = tile * 16 + l16;
        const bool valid = cidx < n_r;
        const int  b     = perm[start + (valid ? cidx : 0)];
        const float* hrow = ent + h_ids[b] * DIM;
        const float* trow = ent + t_ids[b] * DIM;
        v16h bfrag[2];
        #pragma unroll
        for (int kc = 0; kc < 2; ++kc) {
            v16h f;
            #pragma unroll
            for (int v = 0; v < 8; ++v) {
                const int k0c = iclamp(2 * v + 16 * half + 32 * kc, DIM - 2);
                float2 xh = *(const float2*)(hrow + k0c);
                float2 xt = *(const float2*)(trow + k0c);
                f[2 * v]     = (_Float16)(xh.x - xt.x);   // tail garbage OK: A==0 there
                f[2 * v + 1] = (_Float16)(xh.y - xt.y);
            }
            bfrag[kc] = f;
        }
        wmma_tail(a, bfrag, rr, half, valid, b, out);
    }
}

extern "C" void kernel_launch(void* const* d_in, const int* in_sizes, int n_in,
                              void* d_out, int out_size, void* d_ws, size_t ws_size,
                              hipStream_t stream) {
    const float* ent   = (const float*)d_in[0];
    const float* rel   = (const float*)d_in[1];
    const float* mat   = (const float*)d_in[2];
    const int*   h_ids = (const int*)d_in[3];
    const int*   r_typ = (const int*)d_in[4];
    const int*   t_ids = (const int*)d_in[5];
    float*       out   = (float*)d_out;

    const int n = in_sizes[3];   // BATCH

    int* wsi = (int*)d_ws;
    int* counts  = wsi;
    int* offsets = wsi + 256;
    int* cursors = wsi + 512;
    int* perm    = wsi + 768;

    const size_t permEndB = (size_t)(768 + n) * 4;
    const size_t xpOffB   = (permEndB + 127) & ~(size_t)127;   // 128B-align xp rows
    const bool   staged   = ws_size >= xpOffB + (size_t)n * 128;
    _Float16* xp = (_Float16*)((char*)d_ws + xpOffB);

    zero_k<<<3, 256, 0, stream>>>(wsi);
    hist_k<<<(n + 255) / 256, 256, 0, stream>>>(r_typ, counts, n);
    scan_k<<<1, 32, 0, stream>>>(wsi);
    scatter_k<<<(n + 255) / 256, 256, 0, stream>>>(r_typ, cursors, perm, n);

    dim3 grid(REL_N, 8);
    if (staged) {
        xprep_k<<<(n * 32 + 255) / 256, 256, 0, stream>>>(ent, h_ids, t_ids, perm,
                                                          (unsigned*)xp, n);
        transr_wmma_staged_k<<<grid, 32, 0, stream>>>(rel, mat, perm, counts,
                                                      offsets, xp, out);
    } else {
        transr_wmma_direct_k<<<grid, 32, 0, stream>>>(ent, rel, mat, h_ids, t_ids,
                                                      perm, counts, offsets, out);
    }
}
